// MaskedAttentionHead_11390253269215
// MI455X (gfx1250) — compile-verified, matrix operand order checked by reading, not run on hardware
//
#include <hip/hip_runtime.h>

// ---------------------------------------------------------------------------
// MaskedAttentionHead for MI455X (gfx1250): bf16 WMMA flash attention.
// B=4, S=4096, EMB=1024, HEAD=128.  All matrix math via v_wmma_f32_16x16x32_bf16.
// ---------------------------------------------------------------------------

#define EMB   1024
#define HEAD  128
#define SEQ   4096
#define BATCH 4
#define BS    (BATCH * SEQ)   // 16384 total rows

typedef __attribute__((ext_vector_type(16))) __bf16 bf16x16;
typedef __attribute__((ext_vector_type(8)))  __bf16 bf16x8;
typedef __attribute__((ext_vector_type(8)))  float  f32x8;

__device__ __forceinline__ f32x8 wmma_bf16(bf16x16 a, bf16x16 b, f32x8 c) {
  // D = A(16x32 bf16) * B(32x16 bf16) + C(16x16 f32)
  return __builtin_amdgcn_wmma_f32_16x16x32_bf16(
      /*neg_a=*/false, a, /*neg_b=*/false, b,
      /*c_mod=*/(short)0, c, /*reuse_a=*/false, /*reuse_b=*/false);
}

// Load a 16-element bf16 fragment from two contiguous 8-element (16B) chunks.
// Matches the ISA 16-bit A/B layout: lane holds K = kbase+{0..7} and kbase+{16..23}.
__device__ __forceinline__ bf16x16 ld_frag(const __bf16* lo, const __bf16* hi) {
  bf16x8 a = *(const bf16x8*)lo;
  bf16x8 b = *(const bf16x8*)hi;
  return __builtin_shufflevector(a, b, 0,1,2,3,4,5,6,7,8,9,10,11,12,13,14,15);
}

// ---------------------------------------------------------------------------
// Kernel 1: repack Wq/Wk/Wv (f32 [1024][128]) into bf16 B-fragment layout:
//   packW[m][kstep(32)][ntile(8)][lane(32)][elem(16)]
// so the GEMM loads each B fragment as two contiguous 16B chunks per lane.
// ---------------------------------------------------------------------------
__global__ void __launch_bounds__(256) pack_w_kernel(
    const float* __restrict__ Wq, const float* __restrict__ Wk,
    const float* __restrict__ Wv, __bf16* __restrict__ packW)
{
  int idx = blockIdx.x * 256 + threadIdx.x;      // < 3*32*8*32*16 = 393216
  int e    = idx & 15;
  int lane = (idx >> 4) & 31;
  int nt   = (idx >> 9) & 7;
  int ks   = (idx >> 12) & 31;
  int mm   = idx >> 17;
  int k = ks * 32 + ((lane >> 4) * 8) + (e < 8 ? e : e + 8);
  int n = nt * 16 + (lane & 15);
  const float* W = (mm == 0) ? Wq : (mm == 1) ? Wk : Wv;
  packW[idx] = (__bf16)W[(size_t)k * HEAD + n];
}

// ---------------------------------------------------------------------------
// Kernel 2: QKV projection. Each block: 16 rows of X; 8 waves = 8 column tiles
// of 16 (covering HEAD=128); each wave accumulates Q,K,V tiles over K=1024
// with 3 WMMAs per 32-wide K step. Q is pre-scaled by 1/sqrt(HEAD).
// ---------------------------------------------------------------------------
__global__ void __launch_bounds__(256) qkv_proj_kernel(
    const float* __restrict__ X, const __bf16* __restrict__ packW,
    const float* __restrict__ bq, const float* __restrict__ bk,
    const float* __restrict__ bv,
    __bf16* __restrict__ Q, __bf16* __restrict__ K, __bf16* __restrict__ V)
{
  __shared__ __align__(16) __bf16 Xs[16][40];   // 16 rows x 32 K (pad to 40)

  const int tid  = threadIdx.x;
  const int wave = tid >> 5;        // 0..7 -> column tile
  const int lane = tid & 31;
  const int ln15 = lane & 15;
  const int kb   = (lane >> 4) * 8; // kbase within 32-wide K slice
  const int row0 = blockIdx.x * 16;

  f32x8 accQ = {}, accK = {}, accV = {};

  for (int ks = 0; ks < 32; ++ks) {
    // cooperative: 16x32 f32 tile of X -> bf16 LDS (2 elems / thread)
    {
      int r = tid >> 4;             // 0..15
      int c = (tid & 15) * 2;       // 0..30
      const float* src = X + (size_t)(row0 + r) * EMB + ks * 32 + c;
      Xs[r][c]     = (__bf16)src[0];
      Xs[r][c + 1] = (__bf16)src[1];
    }
    __syncthreads();

    bf16x16 afrag = ld_frag(&Xs[ln15][kb], &Xs[ln15][kb + 16]);

    const __bf16* pw = packW + ((size_t)ks * 8 + wave) * (32 * 16) + lane * 16;
    const size_t mstride = (size_t)32 * 8 * 32 * 16;  // per-matrix stride
    bf16x16 bqf = ld_frag(pw,               pw + 8);
    bf16x16 bkf = ld_frag(pw + mstride,     pw + mstride + 8);
    bf16x16 bvf = ld_frag(pw + 2 * mstride, pw + 2 * mstride + 8);

    accQ = wmma_bf16(afrag, bqf, accQ);
    accK = wmma_bf16(afrag, bkf, accK);
    accV = wmma_bf16(afrag, bvf, accV);
    __syncthreads();
  }

  // Epilogue: bias, scale Q, store bf16 row-major [BS][HEAD].
  const int   n   = wave * 16 + ln15;
  const float bqv = bq[n], bkv = bk[n], bvv = bv[n];
  const float scale = 0.08838834764831845f;   // 1/sqrt(128)
  const int rbase = row0 + ((lane >> 4) * 8); // C layout: lanes>=16 hold M=r+8
#pragma unroll
  for (int r = 0; r < 8; ++r) {
    size_t o = (size_t)(rbase + r) * HEAD + n;
    Q[o] = (__bf16)((accQ[r] + bqv) * scale);
    K[o] = (__bf16)(accK[r] + bkv);
    V[o] = (__bf16)(accV[r] + bvv);
  }
}

// ---------------------------------------------------------------------------
// Kernel 3: causal flash attention. Block = 4 waves = 64 query rows of one
// batch. Per 32-key step: K tile + transposed V tile in LDS; per wave:
// 8 WMMAs (QK^T), online softmax, P->LDS->A-fragment, 8 WMMAs (PV).
// ---------------------------------------------------------------------------
__global__ void __launch_bounds__(128) attn_kernel(
    const __bf16* __restrict__ Q, const __bf16* __restrict__ K,
    const __bf16* __restrict__ V, float* __restrict__ Out)
{
  __shared__ __align__(16) __bf16 Ks[32][136];    // [key][head]
  __shared__ __align__(16) __bf16 Vt[128][40];    // [head][key] (transposed)
  __shared__ __align__(16) __bf16 Pb[4][16][40];  // per-wave P staging

  const int tid  = threadIdx.x;
  const int wave = tid >> 5;
  const int lane = tid & 31;
  const int ln15 = lane & 15;
  const int kb   = (lane >> 4) * 8;

  const int b    = blockIdx.x >> 6;     // 64 blocks per batch
  const int q0b  = (blockIdx.x & 63) * 64;
  const int q0w  = q0b + wave * 16;
  const size_t rowbase = (size_t)b * SEQ;

  // Preload Q fragments for this wave's 16 rows (4 slices of K-dim 32).
  bf16x16 qf[4];
  {
    const __bf16* qrow = Q + (rowbase + q0w + ln15) * HEAD;
#pragma unroll
    for (int kk = 0; kk < 4; ++kk)
      qf[kk] = ld_frag(qrow + kk * 32 + kb, qrow + kk * 32 + kb + 16);
  }

  f32x8 Oa[8];
#pragma unroll
  for (int t = 0; t < 8; ++t) Oa[t] = (f32x8){};
  float m[8], l[8];
#pragma unroll
  for (int r = 0; r < 8; ++r) { m[r] = -__builtin_inff(); l[r] = 0.0f; }

  const int jmax = (q0b + 63) >> 5;
  for (int j = 0; j <= jmax; ++j) {
    __syncthreads();                       // protect LDS tiles from prev iter
    // Cooperative load: K row-major, V transposed. 1 key-row segment / thread.
    {
      int key = tid >> 2;                  // 0..31
      int seg = (tid & 3) * 32;            // 0,32,64,96
      const __bf16* krow = K + (rowbase + (size_t)j * 32 + key) * HEAD + seg;
      const __bf16* vrow = V + (rowbase + (size_t)j * 32 + key) * HEAD + seg;
#pragma unroll
      for (int c = 0; c < 4; ++c)
        *(bf16x8*)&Ks[key][seg + 8 * c] = *(const bf16x8*)(krow + 8 * c);
#pragma unroll
      for (int c = 0; c < 4; ++c) {
        bf16x8 v = *(const bf16x8*)(vrow + 8 * c);
#pragma unroll
        for (int i = 0; i < 8; ++i) Vt[seg + 8 * c + i][key] = v[i];
      }
    }
    __syncthreads();

    if (32 * j > q0w + 15) continue;       // fully-masked for this wave

    // ---- scores: S(16x32) = Q(16x128) * K^T, as two 16x16 tiles -----------
    f32x8 s0 = {}, s1 = {};
#pragma unroll
    for (int kk = 0; kk < 4; ++kk) {
      bf16x16 b0 = ld_frag(&Ks[ln15][kk * 32 + kb], &Ks[ln15][kk * 32 + kb + 16]);
      bf16x16 b1 = ld_frag(&Ks[16 + ln15][kk * 32 + kb], &Ks[16 + ln15][kk * 32 + kb + 16]);
      s0 = wmma_bf16(qf[kk], b0, s0);
      s1 = wmma_bf16(qf[kk], b1, s1);
    }

    // ---- causal mask (only boundary tiles) --------------------------------
    const int qrow0 = q0w + ((lane >> 4) * 8);   // row of VGPR r = qrow0 + r
    if (32 * j + 31 > q0w) {
      int key0 = 32 * j + ln15;
#pragma unroll
      for (int r = 0; r < 8; ++r) {
        int q = qrow0 + r;
        if (key0 > q)      s0[r] = -__builtin_inff();
        if (key0 + 16 > q) s1[r] = -__builtin_inff();
      }
    }

    // ---- online softmax ---------------------------------------------------
    float mnew[8], alpha[8];
#pragma unroll
    for (int r = 0; r < 8; ++r) {
      float t = fmaxf(s0[r], s1[r]);
      t = fmaxf(t, __shfl_xor(t, 1));
      t = fmaxf(t, __shfl_xor(t, 2));
      t = fmaxf(t, __shfl_xor(t, 4));
      t = fmaxf(t, __shfl_xor(t, 8));      // row max within 16-lane half
      mnew[r]  = fmaxf(m[r], t);
      alpha[r] = __expf(m[r] - mnew[r]);
    }
    f32x8 p0, p1;
#pragma unroll
    for (int r = 0; r < 8; ++r) {
      p0[r] = __expf(s0[r] - mnew[r]);
      p1[r] = __expf(s1[r] - mnew[r]);
      float rs = p0[r] + p1[r];
      rs += __shfl_xor(rs, 1);
      rs += __shfl_xor(rs, 2);
      rs += __shfl_xor(rs, 4);
      rs += __shfl_xor(rs, 8);
      l[r] = l[r] * alpha[r] + rs;
      m[r] = mnew[r];
    }
#pragma unroll
    for (int t = 0; t < 8; ++t)
#pragma unroll
      for (int r = 0; r < 8; ++r) Oa[t][r] *= alpha[r];

    // ---- P: C-layout regs -> LDS (row-major 16x32 bf16) -> A fragment -----
    {
      int prow0 = (lane >> 4) * 8;
#pragma unroll
      for (int r = 0; r < 8; ++r) {
        Pb[wave][prow0 + r][ln15]      = (__bf16)p0[r];
        Pb[wave][prow0 + r][16 + ln15] = (__bf16)p1[r];
      }
    }
    asm volatile("s_wait_dscnt 0" ::: "memory");   // wave-local store->load
    bf16x16 pf = ld_frag(&Pb[wave][ln15][kb], &Pb[wave][ln15][kb + 16]);

    // ---- O += P(16x32) * V(32x128), 8 column tiles ------------------------
#pragma unroll
    for (int t = 0; t < 8; ++t) {
      bf16x16 vb = ld_frag(&Vt[16 * t + ln15][kb], &Vt[16 * t + ln15][kb + 16]);
      Oa[t] = wmma_bf16(pf, vb, Oa[t]);
    }
  }

  // ---- epilogue: normalize by l, store f32 --------------------------------
  float rl[8];
#pragma unroll
  for (int r = 0; r < 8; ++r) rl[r] = 1.0f / l[r];
  const int rbase = q0w + ((lane >> 4) * 8);
#pragma unroll
  for (int t = 0; t < 8; ++t)
#pragma unroll
    for (int r = 0; r < 8; ++r) {
      size_t o = (rowbase + rbase + r) * HEAD + 16 * t + ln15;
      Out[o] = Oa[t][r] * rl[r];
    }
}

// ---------------------------------------------------------------------------
extern "C" void kernel_launch(void* const* d_in, const int* in_sizes, int n_in,
                              void* d_out, int out_size, void* d_ws, size_t ws_size,
                              hipStream_t stream) {
  (void)in_sizes; (void)n_in; (void)out_size; (void)ws_size;
  const float* X  = (const float*)d_in[0];
  const float* Wq = (const float*)d_in[1];
  const float* bq = (const float*)d_in[2];
  const float* Wk = (const float*)d_in[3];
  const float* bk = (const float*)d_in[4];
  const float* Wv = (const float*)d_in[5];
  const float* bv = (const float*)d_in[6];
  float* Out = (float*)d_out;

  // Workspace layout (bytes): Q | K | V (bf16, 4MB each) | packed weights.
  char* ws = (char*)d_ws;
  __bf16* Qb = (__bf16*)(ws);
  __bf16* Kb = (__bf16*)(ws + 4194304);
  __bf16* Vb = (__bf16*)(ws + 8388608);
  __bf16* Pw = (__bf16*)(ws + 12582912);   // 3*32*8*32*16 bf16 = 768 KB

  pack_w_kernel<<<1536, 256, 0, stream>>>(Wq, Wk, Wv, Pw);
  qkv_proj_kernel<<<BS / 16, 256, 0, stream>>>(X, Pw, bq, bk, bv, Qb, Kb, Vb);
  attn_kernel<<<BATCH * (SEQ / 64), 128, 0, stream>>>(Qb, Kb, Vb, Out);
}
